// HybRNN_79894981640779
// MI455X (gfx1250) — matrix operand order repaired and checked by reading, not compiled, not run on hardware
//
#include <hip/hip_runtime.h>
#include <hip/hip_bf16.h>

typedef __attribute__((ext_vector_type(16))) _Float16 v16h;
typedef __attribute__((ext_vector_type(8)))  _Float16 v8h;
typedef __attribute__((ext_vector_type(8)))  float    v8f;

#define T_STEPS 512
#define BATCH   4096
#define HID     128
#define NACT    8
#define IN_V    138
#define IN_H    144
#define AV_STRIDE 160   // padded K (5 chunks of 32)

// ---- LDS byte offsets (all 16B aligned) ----
#define OFF_WFV   0        // W_hv.T B-frags: 8 ntiles * 5 kchunks * 32 lanes * 16 halfs = 40960 B
#define OFF_WFH   40960    // W_hh.T B-frags: 40960 B
#define OFF_WFU   81920    // W_ru.T B-frags: 4*32*16 halfs = 4096 B
#define OFF_WFL   86016    // W_hl.T B-frags: 4096 B
#define OFF_WFB   90112    // W_rb.T B-frag : 32*16 halfs = 1024 B
#define OFF_AV    91136    // inp_v staging (16 x 160 f16): [chosen,r,value(8),pad(6),v_state(128),pad(16)]
#define OFF_AH    96256    // inp_h staging (16 x 160 f16): [prev_a(8),habit(8),h_state(128),pad(16)]
#define OFF_AVAL  101376   // value staging for nv-WMMA (16 x 32 f16)
#define OFF_NVWS  102400   // nv result (16x16 f32)
#define OFF_HBWS  103424   // habit result (16x16 f32)
#define OFF_RUS   104448   // reward_update (16 f32)
#define OFF_BHV   104512   // b_hv (128 f32)
#define OFF_BHH   105024   // b_hh (128 f32)
#define OFF_BHL   105536   // b_hl (16 f32, padded)
#define OFF_BRB   105600   // b_rb (16 f32, padded)
#define OFF_BRU   105664   // b_ru (1 f32)
#define SMEM_BYTES 105728

// ---- output offsets: logits, hf, vf, habitf, valuef ----
#define OFF_HF  ((size_t)T_STEPS * BATCH * NACT)
#define OFF_VF  (OFF_HF + (size_t)BATCH * HID)
#define OFF_HB  (OFF_VF + (size_t)BATCH * HID)
#define OFF_VL  (OFF_HB + (size_t)BATCH * NACT)

// Load a 16x32 f16 A-fragment from row-major LDS (contiguous K layout:
// lanes 0-15 hold M=lane, halfs K=[0..15]; lanes 16-31 hold K=[16..31]).
__device__ __forceinline__ v16h load_afrag(const _Float16* Abuf, int stride, int kb, int lane) {
  int m    = lane & 15;
  int koff = (lane >> 4) << 4;
  const _Float16* p = Abuf + m * stride + kb + koff;
  v8h lo = *(const v8h*)(p);
  v8h hi = *(const v8h*)(p + 8);
  return __builtin_shufflevector(lo, hi, 0,1,2,3,4,5,6,7,8,9,10,11,12,13,14,15);
}

__device__ __forceinline__ v8f wmma16(v16h a, v16h b, v8f c) {
  return __builtin_amdgcn_wmma_f32_16x16x32_f16(false, a, false, b, (short)0, c, false, false);
}

// Fast tanh on the TRANS pipe: tanh(x) = 1 - 2/(e^{2x}+1).
// v_exp_f32 co-executes with VALU on CDNA5, much shorter critical path than OCML tanh.
__device__ __forceinline__ float fast_tanh(float x) {
  float e = __expf(2.0f * x);
  return 1.0f - 2.0f / (e + 1.0f);
}

__global__ __launch_bounds__(128) void hybrnn_kernel(
    const float* __restrict__ inputs,
    const float* __restrict__ W_rb, const float* __restrict__ b_rb,
    const float* __restrict__ W_hv, const float* __restrict__ b_hv,
    const float* __restrict__ W_ru, const float* __restrict__ b_ru,
    const float* __restrict__ W_hh, const float* __restrict__ b_hh,
    const float* __restrict__ W_hl, const float* __restrict__ b_hl,
    float* __restrict__ out)
{
  extern __shared__ __align__(16) char smem[];
  _Float16* sWfV  = (_Float16*)(smem + OFF_WFV);
  _Float16* sWfH  = (_Float16*)(smem + OFF_WFH);
  _Float16* sWfU  = (_Float16*)(smem + OFF_WFU);
  _Float16* sWfL  = (_Float16*)(smem + OFF_WFL);
  _Float16* sWfB  = (_Float16*)(smem + OFF_WFB);
  _Float16* sAv   = (_Float16*)(smem + OFF_AV);
  _Float16* sAh   = (_Float16*)(smem + OFF_AH);
  _Float16* sAval = (_Float16*)(smem + OFF_AVAL);
  float*    sNvWs = (float*)(smem + OFF_NVWS);
  float*    sHbWs = (float*)(smem + OFF_HBWS);
  float*    sRu   = (float*)(smem + OFF_RUS);
  float*    sBhv  = (float*)(smem + OFF_BHV);
  float*    sBhh  = (float*)(smem + OFF_BHH);
  float*    sBhl  = (float*)(smem + OFF_BHL);
  float*    sBrb  = (float*)(smem + OFF_BRB);
  float*    sBru  = (float*)(smem + OFF_BRU);

  const int tid   = threadIdx.x;
  const int lane  = tid & 31;
  const int wave  = tid >> 5;
  const int rbase = blockIdx.x * 16;

  // ---------- one-time weight -> fragment conversion (f32 -> f16) ----------
  // Big GEMM B-frags: frag[((nt*5+kc)*32+lane)*16+j] = B[k][n], n=nt*16+(lane&15),
  // k = kc*32 + (lane>>4)*16 + j  (B = W.T with padded/remapped K rows).
  for (int idx = tid; idx < 8 * 5 * 32 * 16; idx += 128) {
    int j = idx & 15, ln = (idx >> 4) & 31, kcnt = idx >> 9;
    int kc = kcnt % 5, nt = kcnt / 5;
    int n = nt * 16 + (ln & 15);
    int k = kc * 32 + ((ln >> 4) << 4) + j;
    float wv = 0.f, wh = 0.f;
    if (k < 10)                  wv = W_hv[n * IN_V + k];        // chosen,r,value cols
    else if (k >= 16 && k < 144) wv = W_hv[n * IN_V + (k - 6)];  // v_state cols (realigned)
    if (k < IN_H)                wh = W_hh[n * IN_H + k];
    sWfV[idx] = (_Float16)wv;
    sWfH[idx] = (_Float16)wh;
  }
  for (int idx = tid; idx < 4 * 32 * 16; idx += 128) {
    int j = idx & 15, ln = (idx >> 4) & 31, kc = idx >> 9;
    int n = ln & 15;
    int k = kc * 32 + ((ln >> 4) << 4) + j;   // k < 128
    sWfU[idx] = (_Float16)((n == 0) ? W_ru[k] : 0.f);
    sWfL[idx] = (_Float16)((n < 8) ? W_hl[n * HID + k] : 0.f);
  }
  for (int idx = tid; idx < 32 * 16; idx += 128) {
    int j = idx & 15, ln = idx >> 4;
    int n = ln & 15;
    int k = ((ln >> 4) << 4) + j;
    sWfB[idx] = (_Float16)((n < 8 && k < 8) ? W_rb[n * 8 + k] : 0.f);
  }
  // State init: value = 0.5, everything else 0
  for (int idx = tid; idx < 16 * AV_STRIDE; idx += 128) {
    int c = idx % AV_STRIDE;
    sAv[idx] = (_Float16)((c >= 2 && c < 10) ? 0.5f : 0.f);
    sAh[idx] = (_Float16)0.f;
  }
  for (int idx = tid; idx < 16 * 32; idx += 128)
    sAval[idx] = (_Float16)(((idx & 31) < 8) ? 0.5f : 0.f);
  if (tid < 128) { sBhv[tid] = b_hv[tid]; sBhh[tid] = b_hh[tid]; }
  if (tid < 16) {
    sBhl[tid] = (tid < 8) ? b_hl[tid] : 0.f;
    sBrb[tid] = (tid < 8) ? b_rb[tid] : 0.f;
  }
  if (tid == 0) sBru[0] = b_ru[0];
  __syncthreads();

  const int row = tid >> 3, col = tid & 7;  // 16x8 element ownership
  const int n_c = lane & 15;                // C/D tile column
  const int mh  = (lane >> 4) << 3;         // C/D tile row base

  for (int t = 0; t < T_STEPS; ++t) {
    // ---- phase a: stream inputs; prefetch next step's slice ----
    float a_val = inputs[((size_t)t * BATCH + (rbase + row)) * 9 + col];
    if (t + 1 < T_STEPS)
      __builtin_prefetch(&inputs[((size_t)(t + 1) * BATCH + (rbase + row)) * 9 + col], 0, 0);
    if (tid < 16) {
      float r = inputs[((size_t)t * BATCH + (rbase + tid)) * 9 + 8];
      sAv[tid * AV_STRIDE + 1] = (_Float16)r;
    }
    // ---- phase b: nv = value @ W_rb.T + b_rb (wave 0, one WMMA) ----
    if (wave == 0) {
      v16h a = load_afrag(sAval, 32, 0, lane);
      v16h b = *(const v16h*)(sWfB + lane * 16);
      v8f c; float bv = sBrb[n_c];
      #pragma unroll
      for (int i = 0; i < 8; ++i) c[i] = bv;
      c = wmma16(a, b, c);
      #pragma unroll
      for (int i = 0; i < 8; ++i) sNvWs[(mh + i) * 16 + n_c] = c[i];
    }
    __syncthreads();
    // ---- phase c: chosen = sum(nv * a) via 8-lane xor-shuffle reduce ----
    float nv0 = sNvWs[row * 16 + col];
    float ch = nv0 * a_val;
    ch += __shfl_xor(ch, 1, 32);
    ch += __shfl_xor(ch, 2, 32);
    ch += __shfl_xor(ch, 4, 32);
    if (col == 0) sAv[row * AV_STRIDE + 0] = (_Float16)ch;
    __syncthreads();
    // ---- phase d: V-GEMM (waves 0,1) and H-GEMM (waves 2,3), tanh ----
    const _Float16* Abuf  = (wave < 2) ? sAv  : sAh;
    const _Float16* Wfrag = (wave < 2) ? sWfV : sWfH;
    const float*    bias  = (wave < 2) ? sBhv : sBhh;
    const int ntbase = (wave & 1) * 4;
    v16h af[5];
    #pragma unroll
    for (int kc = 0; kc < 5; ++kc) af[kc] = load_afrag(Abuf, AV_STRIDE, kc * 32, lane);
    v8f ct[4];
    #pragma unroll
    for (int tt = 0; tt < 4; ++tt) {
      int nt = ntbase + tt;
      float bv = bias[nt * 16 + n_c];
      v8f c;
      #pragma unroll
      for (int i = 0; i < 8; ++i) c[i] = bv;
      #pragma unroll
      for (int kc = 0; kc < 5; ++kc) {
        v16h b = *(const v16h*)(Wfrag + ((nt * 5 + kc) * 32 + lane) * 16);
        c = wmma16(af[kc], b, c);
      }
      #pragma unroll
      for (int i = 0; i < 8; ++i) c[i] = fast_tanh(c[i]);
      ct[tt] = c;
    }
    __syncthreads();
    // ---- phase e: commit n_vstate/n_hstate (f16) back into staging ----
    {
      _Float16* Sbuf = (wave < 2) ? sAv : sAh;
      #pragma unroll
      for (int tt = 0; tt < 4; ++tt) {
        int cb = 16 + (ntbase + tt) * 16 + n_c;
        #pragma unroll
        for (int i = 0; i < 8; ++i) Sbuf[(mh + i) * AV_STRIDE + cb] = (_Float16)ct[tt][i];
      }
      if (t == T_STEPS - 1) {  // final vf / hf carries (f32)
        float* dst = out + ((wave < 2) ? OFF_VF : OFF_HF);
        #pragma unroll
        for (int tt = 0; tt < 4; ++tt)
          #pragma unroll
          for (int i = 0; i < 8; ++i)
            dst[(size_t)(rbase + mh + i) * HID + (ntbase + tt) * 16 + n_c] = ct[tt][i];
      }
    }
    __syncthreads();
    // ---- phase f: reward_update (wave 1) and n_habit (wave 2) WMMAs ----
    if (wave == 1) {
      v8f c; float bv = (n_c == 0) ? sBru[0] : 0.f;
      #pragma unroll
      for (int i = 0; i < 8; ++i) c[i] = bv;
      #pragma unroll
      for (int kc = 0; kc < 4; ++kc) {
        v16h a = load_afrag(sAv + 16, AV_STRIDE, kc * 32, lane);
        v16h b = *(const v16h*)(sWfU + (kc * 32 + lane) * 16);
        c = wmma16(a, b, c);
      }
      if (n_c == 0) {
        #pragma unroll
        for (int i = 0; i < 8; ++i) sRu[mh + i] = c[i];
      }
    } else if (wave == 2) {
      v8f c; float bv = sBhl[n_c];
      #pragma unroll
      for (int i = 0; i < 8; ++i) c[i] = bv;
      #pragma unroll
      for (int kc = 0; kc < 4; ++kc) {
        v16h a = load_afrag(sAh + 16, AV_STRIDE, kc * 32, lane);
        v16h b = *(const v16h*)(sWfL + (kc * 32 + lane) * 16);
        c = wmma16(a, b, c);
      }
      #pragma unroll
      for (int i = 0; i < 8; ++i) sHbWs[(mh + i) * 16 + n_c] = c[i];
    }
    __syncthreads();
    // ---- phase g: nv += a*ru + habit; emit logits; roll carries ----
    float habit = sHbWs[row * 16 + col];
    float nv = nv0 + a_val * sRu[row] + habit;
    out[((size_t)t * BATCH + (rbase + row)) * NACT + col] = nv;
    sAv[row * AV_STRIDE + 2 + col]  = (_Float16)nv;     // value
    sAval[row * 32 + col]           = (_Float16)nv;
    sAh[row * AV_STRIDE + col]      = (_Float16)a_val;  // prev_a
    sAh[row * AV_STRIDE + 8 + col]  = (_Float16)habit;  // habit
    if (t == T_STEPS - 1) {
      out[OFF_HB + (size_t)(rbase + row) * NACT + col] = habit;
      out[OFF_VL + (size_t)(rbase + row) * NACT + col] = nv;
    }
    __syncthreads();
  }
}

extern "C" void kernel_launch(void* const* d_in, const int* in_sizes, int n_in,
                              void* d_out, int out_size, void* d_ws, size_t ws_size,
                              hipStream_t stream) {
  (void)in_sizes; (void)n_in; (void)out_size; (void)d_ws; (void)ws_size;
  const float* inputs = (const float*)d_in[0];
  const float* W_rb   = (const float*)d_in[1];
  const float* b_rb   = (const float*)d_in[2];
  const float* W_hv   = (const float*)d_in[3];
  const float* b_hv   = (const float*)d_in[4];
  const float* W_ru   = (const float*)d_in[5];
  const float* b_ru   = (const float*)d_in[6];
  const float* W_hh   = (const float*)d_in[7];
  const float* b_hh   = (const float*)d_in[8];
  const float* W_hl   = (const float*)d_in[9];
  const float* b_hl   = (const float*)d_in[10];
  float* out = (float*)d_out;

  dim3 grid(BATCH / 16), block(128);
  hybrnn_kernel<<<grid, block, SMEM_BYTES, stream>>>(
      inputs, W_rb, b_rb, W_hv, b_hv, W_ru, b_ru, W_hh, b_hh, W_hl, b_hl, out);
}
